// CrossAttention_29008209117259
// MI455X (gfx1250) — matrix operand order repaired and checked by reading, not compiled, not run on hardware
//
#include <hip/hip_runtime.h>

#define BB 16
#define LQ 64
#define LD 2048
#define DD 256

typedef __attribute__((ext_vector_type(16))) _Float16 v16h;
typedef __attribute__((ext_vector_type(8)))  float    v8f;

// WMMA operand layouts (CDNA5 ISA 7.12.2, wave32):
//  A (16x32 f16): lane<->M (m = lane%16); regs: e=0..7 -> K = 8*hi + e,
//                 e=8..15 -> K = 16 + 8*hi + e-8   (hi = lane/16)
//  B (32x16 f16): lane<->N (n = lane%16); regs as A; code feeds A and B with K = 16*g + e (g = lane/16)
//  C/D (16x16 f32): reg r, lane l -> (M = 8*(l/16) + r, N = l%16)
__device__ __forceinline__ v8f wmma_f16(v16h a, v16h b, v8f c) {
  return __builtin_amdgcn_wmma_f32_16x16x32_f16(false, a, false, b, (short)0, c,
                                                false, false);
}

__device__ __forceinline__ v8f vzero8() {
  v8f z;
#pragma unroll
  for (int e = 0; e < 8; ++e) z[e] = 0.0f;
  return z;
}

// ---------------------------------------------------------------------------
// K0: qw2[b,j] = query[b,j,:]·w2 ; f16 query row-major (q_h) and d-major (q_t).
__global__ void __launch_bounds__(256) k_prep_q(const float* __restrict__ query,
                                                const float* __restrict__ W,
                                                float* __restrict__ qw2,
                                                _Float16* __restrict__ q_h,
                                                _Float16* __restrict__ q_t) {
  int b = blockIdx.x / LQ, j = blockIdx.x % LQ, t = threadIdx.x;
  __shared__ float red[256];
  float qv = query[(size_t)(b * LQ + j) * DD + t];
  red[t] = qv * W[DD + t];
  __syncthreads();
  for (int o = 128; o > 0; o >>= 1) {
    if (t < o) red[t] += red[t + o];
    __syncthreads();
  }
  if (t == 0) qw2[b * LQ + j] = red[0];
  q_h[(size_t)(b * LQ + j) * DD + t] = (_Float16)qv;
  q_t[((size_t)b * DD + t) * LQ + j] = (_Float16)qv;
}

// ---------------------------------------------------------------------------
// K1: dw1[b,i] = doc[b,i,:]·w1
__global__ void __launch_bounds__(256) k_prep_d(const float* __restrict__ doc,
                                                const float* __restrict__ W,
                                                float* __restrict__ dw1) {
  int b = blockIdx.x / LD, i = blockIdx.x % LD, t = threadIdx.x;
  __shared__ float red[256];
  red[t] = doc[(size_t)(b * LD + i) * DD + t] * W[t];
  __syncthreads();
  for (int o = 128; o > 0; o >>= 1) {
    if (t < o) red[t] += red[t + o];
    __syncthreads();
  }
  if (t == 0) dw1[b * LD + i] = red[0];
}

// ---------------------------------------------------------------------------
// K2: S[b,i,j] = (doc⊙w3)@q^T + dw1[i] + qw2[j]  via WMMA f16 / f32 acc.
// Block = 128 = 4 waves; each wave owns one 16-row m-tile x all 64 j.
__global__ void __launch_bounds__(128) k_score(const float* __restrict__ doc,
                                               const float* __restrict__ W,
                                               const _Float16* __restrict__ q_h,
                                               const float* __restrict__ dw1,
                                               const float* __restrict__ qw2,
                                               float* __restrict__ S) {
  int b = blockIdx.x / (LD / 64);
  int tile = blockIdx.x % (LD / 64);
  int wave = threadIdx.x >> 5, lane = threadIdx.x & 31;
  int m = lane & 15, hi = lane >> 4;  // A: row m, K-chunk hi.  B: col m, K-group hi.
  int i0 = tile * 64 + wave * 16;
  const float* w3 = W + 2 * DD;
  const float* drow = doc + (size_t)(b * LD + i0 + m) * DD;
  v8f acc[4];
#pragma unroll
  for (int nt = 0; nt < 4; ++nt) acc[nt] = vzero8();

  for (int ks = 0; ks < 8; ++ks) {  // K = D = 256, 32 per WMMA
    int d0 = ks * 32;
    v16h a;
#pragma unroll
    for (int e = 0; e < 8; ++e) {
      int d1 = d0 + hi * 16 + e;
      int d2 = d0 + hi * 16 + 8 + e;
      a[e] = (_Float16)(drow[d1] * w3[d1]);
      a[e + 8] = (_Float16)(drow[d2] * w3[d2]);
    }
#pragma unroll
    for (int nt = 0; nt < 4; ++nt) {
      // B[k=d][n=j]: lane holds column j = nt*16 + m, regs = d0 + hi*16 + e
      v16h bt = *(const v16h*)(q_h + (size_t)(b * LQ + nt * 16 + m) * DD +
                               d0 + hi * 16);
      acc[nt] = wmma_f16(a, bt, acc[nt]);
    }
  }
  const float* dwp = dw1 + b * LD + i0 + hi * 8;
  float dwv[8];
#pragma unroll
  for (int r = 0; r < 8; ++r) dwv[r] = dwp[r];
#pragma unroll
  for (int nt = 0; nt < 4; ++nt) {
    float q2 = qw2[b * LQ + nt * 16 + m];
#pragma unroll
    for (int r = 0; r < 8; ++r) {
      int i = i0 + hi * 8 + r;
      S[(size_t)(b * LD + i) * LQ + nt * 16 + m] = acc[nt][r] + dwv[r] + q2;
    }
  }
}

// ---------------------------------------------------------------------------
// K3: row softmax (over j=64) -> P_Q2D stored TRANSPOSED f16 [b][j][i]
__global__ void __launch_bounds__(64) k_rowsoft(const float* __restrict__ S,
                                                _Float16* __restrict__ Pq2dT) {
  int b = blockIdx.x / LD, i = blockIdx.x % LD, t = threadIdx.x;
  __shared__ float red[64];
  float s = S[(size_t)(b * LD + i) * LQ + t];
  red[t] = s;
  __syncthreads();
  for (int o = 32; o > 0; o >>= 1) {
    if (t < o) red[t] = fmaxf(red[t], red[t + o]);
    __syncthreads();
  }
  float mx = red[0];
  __syncthreads();
  float e = __expf(s - mx);
  red[t] = e;
  __syncthreads();
  for (int o = 32; o > 0; o >>= 1) {
    if (t < o) red[t] += red[t + o];
    __syncthreads();
  }
  Pq2dT[((size_t)(b * LQ) + t) * LD + i] = (_Float16)(e / red[0]);
}

// ---------------------------------------------------------------------------
// K4: column softmax (over i=2048) -> P_D2Q f16 [b][i][j]
__global__ void __launch_bounds__(256) k_colsoft(const float* __restrict__ S,
                                                 _Float16* __restrict__ Pd2q) {
  int b = blockIdx.x / LQ, j = blockIdx.x % LQ, t = threadIdx.x;
  __shared__ float red[256];
  float v[8];
  float mx = -3.0e38f;
#pragma unroll
  for (int r = 0; r < 8; ++r) {
    v[r] = S[(size_t)(b * LD + r * 256 + t) * LQ + j];
    mx = fmaxf(mx, v[r]);
  }
  red[t] = mx;
  __syncthreads();
  for (int o = 128; o > 0; o >>= 1) {
    if (t < o) red[t] = fmaxf(red[t], red[t + o]);
    __syncthreads();
  }
  mx = red[0];
  __syncthreads();
  float e[8], se = 0.0f;
#pragma unroll
  for (int r = 0; r < 8; ++r) {
    e[r] = __expf(v[r] - mx);
    se += e[r];
  }
  red[t] = se;
  __syncthreads();
  for (int o = 128; o > 0; o >>= 1) {
    if (t < o) red[t] += red[t + o];
    __syncthreads();
  }
  float inv = 1.0f / red[0];
#pragma unroll
  for (int r = 0; r < 8; ++r)
    Pd2q[(size_t)(b * LD + r * 256 + t) * LQ + j] = (_Float16)(e[r] * inv);
}

// ---------------------------------------------------------------------------
// K5: T = P_Q2D^T @ doc  (M=64, K=2048, N=256); result stored TRANSPOSED
// d-major as T_t[b][d][j] so k_out's B-operand loads are contiguous.
__global__ void __launch_bounds__(32) k_tgemm(const _Float16* __restrict__ Pq2dT,
                                              const float* __restrict__ doc,
                                              _Float16* __restrict__ T_t) {
  int b = blockIdx.x >> 3;
  int n0 = (blockIdx.x & 7) * 32;
  int lane = threadIdx.x & 31;
  int m = lane & 15, hi = lane >> 4;
  v8f acc[4][2];
#pragma unroll
  for (int mt = 0; mt < 4; ++mt)
#pragma unroll
    for (int nt = 0; nt < 2; ++nt) acc[mt][nt] = vzero8();

  for (int ks = 0; ks < 64; ++ks) {
    int kb = ks * 32;
    // B[k=i][n=d]: lane holds column d = n0 + nt*16 + m; regs e -> i = kb + hi*16 + e
    v16h bt[2];
#pragma unroll
    for (int nt = 0; nt < 2; ++nt)
#pragma unroll
      for (int e = 0; e < 16; ++e)
        bt[nt][e] = (_Float16)doc[(size_t)(b * LD + kb + hi * 16 + e) * DD +
                                  n0 + nt * 16 + m];
#pragma unroll
    for (int mt = 0; mt < 4; ++mt) {
      // A[m=j][k=i]: lane m holds row j = mt*16 + m of P_Q2D^T (contig in i)
      const _Float16* arow = Pq2dT + (size_t)(b * LQ + mt * 16 + m) * LD + kb;
      v16h a;
#pragma unroll
      for (int e = 0; e < 8; ++e) {
        a[e] = arow[hi * 16 + e];
        a[e + 8] = arow[hi * 16 + 8 + e];
      }
      acc[mt][0] = wmma_f16(a, bt[0], acc[mt][0]);
      acc[mt][1] = wmma_f16(a, bt[1], acc[mt][1]);
    }
  }
#pragma unroll
  for (int mt = 0; mt < 4; ++mt)
#pragma unroll
    for (int nt = 0; nt < 2; ++nt)
#pragma unroll
      for (int r = 0; r < 8; ++r) {
        int j = mt * 16 + hi * 8 + r;       // M
        int d = n0 + nt * 16 + m;           // N
        T_t[(size_t)(b * DD + d) * LQ + j] = (_Float16)acc[mt][nt][r];
      }
}

// ---------------------------------------------------------------------------
// K6: A_D2Q = P_D2Q@q ; A_Q2D = P_D2Q@T (shared A fragments, K=64 -> 2 steps);
// fused epilogue writes concat([doc, A_D2Q, doc*A_D2Q, doc*A_Q2D]).
__global__ void __launch_bounds__(128) k_out(const _Float16* __restrict__ Pd2q,
                                             const _Float16* __restrict__ q_t,
                                             const _Float16* __restrict__ T_t,
                                             const float* __restrict__ doc,
                                             float* __restrict__ out) {
  int b = blockIdx.x / (LD / 64);
  int tile = blockIdx.x % (LD / 64);
  int wave = threadIdx.x >> 5, lane = threadIdx.x & 31;
  int m = lane & 15, hi = lane >> 4;
  int i0 = tile * 64 + wave * 16;
  const _Float16* arow = Pd2q + (size_t)(b * LD + i0 + m) * LQ;

  for (int nb = 0; nb < 4; ++nb) {  // 64-wide d-chunks of N=256
    int dbase = nb * 64;
    v8f acc1[4], acc2[4];
#pragma unroll
    for (int nt = 0; nt < 4; ++nt) {
      acc1[nt] = vzero8();
      acc2[nt] = vzero8();
    }
#pragma unroll
    for (int ks = 0; ks < 2; ++ks) {
      int kb = ks * 32;
      v16h a;
#pragma unroll
      for (int e = 0; e < 8; ++e) {
        a[e] = arow[kb + hi * 16 + e];
        a[e + 8] = arow[kb + hi * 16 + 8 + e];
      }
      int j0 = kb + hi * 16;  // B regs e -> j = j0 + e; lane holds column d
#pragma unroll
      for (int nt = 0; nt < 4; ++nt) {
        int d = dbase + nt * 16 + m;
        v16h b1 = *(const v16h*)(q_t + (size_t)(b * DD + d) * LQ + j0);
        v16h b2 = *(const v16h*)(T_t + (size_t)(b * DD + d) * LQ + j0);
        acc1[nt] = wmma_f16(a, b1, acc1[nt]);
        acc2[nt] = wmma_f16(a, b2, acc2[nt]);
      }
    }
#pragma unroll
    for (int nt = 0; nt < 4; ++nt) {
#pragma unroll
      for (int r = 0; r < 8; ++r) {
        int i = i0 + hi * 8 + r;
        int d = dbase + nt * 16 + m;
        float dv = doc[(size_t)(b * LD + i) * DD + d];
        float a1 = acc1[nt][r], a2 = acc2[nt][r];
        size_t ob = (size_t)(b * LD + i) * (4 * DD);
        out[ob + d] = dv;
        out[ob + DD + d] = a1;
        out[ob + 2 * DD + d] = dv * a1;
        out[ob + 3 * DD + d] = dv * a2;
      }
    }
  }
}

// ---------------------------------------------------------------------------
extern "C" void kernel_launch(void* const* d_in, const int* in_sizes, int n_in,
                              void* d_out, int out_size, void* d_ws, size_t ws_size,
                              hipStream_t stream) {
  (void)in_sizes; (void)n_in; (void)out_size; (void)ws_size;
  const float* query = (const float*)d_in[0];
  const float* doc = (const float*)d_in[1];
  const float* W = (const float*)d_in[2];
  float* out = (float*)d_out;

  char* ws = (char*)d_ws;
  size_t off = 0;
  float* S = (float*)(ws + off);        off += (size_t)BB * LD * LQ * 4;   // 8 MB
  float* dw1 = (float*)(ws + off);      off += (size_t)BB * LD * 4;
  float* qw2 = (float*)(ws + off);      off += (size_t)BB * LQ * 4;
  _Float16* q_t = (_Float16*)(ws + off);   off += (size_t)BB * DD * LQ * 2;
  _Float16* q_h = (_Float16*)(ws + off);   off += (size_t)BB * LQ * DD * 2;
  _Float16* Pd2q = (_Float16*)(ws + off);  off += (size_t)BB * LD * LQ * 2; // 4 MB
  _Float16* Pq2dT = (_Float16*)(ws + off); off += (size_t)BB * LQ * LD * 2; // 4 MB
  _Float16* T_t = (_Float16*)(ws + off);   off += (size_t)BB * DD * LQ * 2;

  k_prep_q<<<BB * LQ, 256, 0, stream>>>(query, W, qw2, q_h, q_t);
  k_prep_d<<<BB * LD, 256, 0, stream>>>(doc, W, dw1);
  k_score<<<BB * (LD / 64), 128, 0, stream>>>(doc, W, q_h, dw1, qw2, S);
  k_rowsoft<<<BB * LD, 64, 0, stream>>>(S, Pq2dT);
  k_colsoft<<<BB * LQ, 256, 0, stream>>>(S, Pd2q);
  k_tgemm<<<BB * 8, 32, 0, stream>>>(Pq2dT, doc, T_t);
  k_out<<<BB * (LD / 64), 128, 0, stream>>>(Pd2q, q_t, T_t, doc, out);
}
